// NeuroGNN_GNN_GraphConv_24773371363442
// MI455X (gfx1250) — compile-verified, matrix operand order checked by reading, not run on hardware
//
#include <hip/hip_runtime.h>

#define NNODES 16384
#define BM 128
#define BN 64
#define BK 32
#define BKP 40   // padded LDS K-stride (multiple of 8 elems -> 16B-aligned v8bf reads)

typedef __attribute__((ext_vector_type(16))) __bf16 v16bf;
typedef __attribute__((ext_vector_type(8)))  __bf16 v8bf;
typedef __attribute__((ext_vector_type(4)))  __bf16 v4bf;
typedef __attribute__((ext_vector_type(2)))  __bf16 v2bf;
typedef __attribute__((ext_vector_type(8)))  float  v8f;
typedef __attribute__((ext_vector_type(4)))  float  v4f;

__device__ __forceinline__ v16bf cat8(v8bf lo, v8bf hi) {
  return __builtin_shufflevector(lo, hi, 0,1,2,3,4,5,6,7,8,9,10,11,12,13,14,15);
}

// 16 bf16 per lane: 8 at p (K = half*8..+7), 8 at p+16 (K = 16+half*8..+7)
__device__ __forceinline__ v16bf load16_bf(const __bf16* __restrict__ p) {
  return cat8(*(const v8bf*)p, *(const v8bf*)(p + 16));
}

// ---------------------------------------------------------------------------
// agg_bf16 = adj^T @ h   (adj fp32 row-major [k][m]; h bf16 row-major [k][f])
// Grid (NNODES/BM, F/BN), 256 threads = 8 waves. Double-buffered LDS; A tile
// transpose-converted fp32->bf16 with packed b32 stores; 4 WMMAs per K-step
// per wave on v_wmma_f32_16x16x32_bf16, fp32 accumulate, bf16 writeback.
// ---------------------------------------------------------------------------
__global__ __launch_bounds__(256) void gemm_aggT(const float* __restrict__ adj,
                                                 const __bf16* __restrict__ hbf,
                                                 __bf16* __restrict__ aggbf, int F) {
  __shared__ __align__(16) __bf16 Alds[2][BM * BKP];  // A[m][k] = adj[k][m]
  __shared__ __align__(16) __bf16 Blds[2][BN * BKP];  // B[n][k] = h[k][n]

  const int tid  = threadIdx.x;
  const int wave = tid >> 5;
  const int lane = tid & 31;
  const int half = lane >> 4;   // K-half select per ISA fragment layout
  const int l16  = lane & 15;   // M (A) / N (B) index within 16x16 tile
  const int m0 = blockIdx.x * BM;
  const int n0 = blockIdx.y * BN;

  // Staging roles: A uses all 256 threads (2 K-rows x 8 M each),
  // B uses threads 0..127 (2 K-rows x 8 N each).
  const int kpA = tid >> 4;   // 0..15 -> K rows 2*kpA, 2*kpA+1
  const int mgA = tid & 15;   // 0..15 -> M cols mgA*8..+7
  const int kpB = tid >> 3;   // 0..15 (tid < 128)
  const int ngB = tid & 7;    // 0..7  -> N cols ngB*8..+7

  v8f acc[4];
#pragma unroll
  for (int t = 0; t < 4; ++t)
#pragma unroll
    for (int r = 0; r < 8; ++r) acc[t][r] = 0.0f;

  auto stage = [&](int buf, int k0) {
    // --- A: 32x128 fp32, coalesced v4f loads, packed-pair bf16 b32 stores ---
    const float* ap0 = adj + (size_t)(k0 + 2 * kpA) * NNODES + m0 + mgA * 8;
    const float* ap1 = ap0 + NNODES;
    __builtin_prefetch(ap0 + (size_t)2 * BK * NNODES, 0, 0);  // 2 tiles ahead
    v4f r0a = ((const v4f*)ap0)[0], r0b = ((const v4f*)ap0)[1];
    v4f r1a = ((const v4f*)ap1)[0], r1b = ((const v4f*)ap1)[1];
#pragma unroll
    for (int j = 0; j < 4; ++j) {
      v2bf p0; p0[0] = (__bf16)r0a[j]; p0[1] = (__bf16)r1a[j];
      *(v2bf*)&Alds[buf][(mgA * 8 + j) * BKP + 2 * kpA] = p0;
      v2bf p1; p1[0] = (__bf16)r0b[j]; p1[1] = (__bf16)r1b[j];
      *(v2bf*)&Alds[buf][(mgA * 8 + 4 + j) * BKP + 2 * kpA] = p1;
    }
    // --- B: 32x64 bf16, v8bf loads, packed-pair b32 transpose stores ---
    if (tid < 128) {
      const __bf16* bp = hbf + (size_t)(k0 + 2 * kpB) * F + n0 + ngB * 8;
      v8bf b0 = *(const v8bf*)bp;
      v8bf b1 = *(const v8bf*)(bp + F);
#pragma unroll
      for (int j = 0; j < 8; ++j) {
        v2bf p; p[0] = b0[j]; p[1] = b1[j];
        *(v2bf*)&Blds[buf][(ngB * 8 + j) * BKP + 2 * kpB] = p;
      }
    }
  };

  const int NT = NNODES / BK;  // 512 K-steps
  stage(0, 0);
  for (int kt = 0; kt < NT; ++kt) {
    const int cur = kt & 1;
    __syncthreads();
    if (kt + 1 < NT) stage(cur ^ 1, (kt + 1) * BK);  // overlap with compute

    const __bf16* Ab = &Alds[cur][(wave * 16 + l16) * BKP + half * 8];
    v16bf a = cat8(*(const v8bf*)Ab, *(const v8bf*)(Ab + 16));
    v16bf b[4];
#pragma unroll
    for (int t = 0; t < 4; ++t) {  // all LDS loads first -> one dscnt wait
      const __bf16* Bb = &Blds[cur][(t * 16 + l16) * BKP + half * 8];
      b[t] = cat8(*(const v8bf*)Bb, *(const v8bf*)(Bb + 16));
    }
#pragma unroll
    for (int t = 0; t < 4; ++t)
      acc[t] = __builtin_amdgcn_wmma_f32_16x16x32_bf16(
          false, a, false, b[t], (short)0, acc[t], false, false);
  }

  // C layout: lanes 0-15 -> M=r, lanes 16-31 -> M=8+r; N = lane&15.
  const int mg = m0 + wave * 16 + half * 8;
#pragma unroll
  for (int t = 0; t < 4; ++t) {
    const int n = n0 + t * 16 + l16;
#pragma unroll
    for (int r = 0; r < 8; ++r)
      aggbf[(size_t)(mg + r) * F + n] = (__bf16)acc[t][r];
  }
}

// ---------------------------------------------------------------------------
// out = relu(agg @ Wrel^T + bias + h @ Wroot^T), all operands bf16 row-major,
// A fragments straight from global (K-contiguous b128), B = weight rows
// (K-contiguous, L2-resident). 8 WMMAs per K-step per wave. Output bf16 for
// inner layers, fp32 for the final layer (template).
// ---------------------------------------------------------------------------
template <bool F32OUT>
__global__ __launch_bounds__(256) void combine_relu(const __bf16* __restrict__ agg,
                                                    const __bf16* __restrict__ h,
                                                    const __bf16* __restrict__ Wrel,
                                                    const __bf16* __restrict__ Wroot,
                                                    const float* __restrict__ bias,
                                                    void* __restrict__ outp, int Fin) {
  const int wave = threadIdx.x >> 5;
  const int lane = threadIdx.x & 31;
  const int half = lane >> 4;
  const int l16  = lane & 15;
  const int mb = (blockIdx.x * 8 + wave) * 16;

  v8f acc[4];
#pragma unroll
  for (int t = 0; t < 4; ++t)
#pragma unroll
    for (int r = 0; r < 8; ++r) acc[t][r] = 0.0f;

  for (int k0 = 0; k0 < Fin; k0 += 32) {
    const size_t arow = (size_t)(mb + l16) * Fin + k0 + half * 8;
    v16bf aAgg = load16_bf(agg + arow);
    v16bf aH   = load16_bf(h + arow);
#pragma unroll
    for (int t = 0; t < 4; ++t) {
      const size_t wrow = (size_t)(t * 16 + l16) * Fin + k0 + half * 8;
      v16bf brel  = load16_bf(Wrel + wrow);
      v16bf broot = load16_bf(Wroot + wrow);
      acc[t] = __builtin_amdgcn_wmma_f32_16x16x32_bf16(
          false, aAgg, false, brel, (short)0, acc[t], false, false);
      acc[t] = __builtin_amdgcn_wmma_f32_16x16x32_bf16(
          false, aH, false, broot, (short)0, acc[t], false, false);
    }
  }

#pragma unroll
  for (int t = 0; t < 4; ++t) {
    const int n = t * 16 + l16;
    const float bv = bias[n];
#pragma unroll
    for (int r = 0; r < 8; ++r) {
      float v = acc[t][r] + bv;
      v = v > 0.0f ? v : 0.0f;
      const size_t idx = (size_t)(mb + half * 8 + r) * 64 + n;
      if (F32OUT)
        ((float*)outp)[idx] = v;
      else
        ((__bf16*)outp)[idx] = (__bf16)v;
    }
  }
}

// fp32 -> bf16 elementwise (n multiple of 4); used for X and the six weights.
__global__ void cvt_f32_to_bf16(const float* __restrict__ in, __bf16* __restrict__ out,
                                int n4) {
  int i = blockIdx.x * blockDim.x + threadIdx.x;
  if (i < n4) {
    v4f f = ((const v4f*)in)[i];
    v4bf o;
#pragma unroll
    for (int j = 0; j < 4; ++j) o[j] = (__bf16)f[j];
    ((v4bf*)out)[i] = o;
  }
}

extern "C" void kernel_launch(void* const* d_in, const int* in_sizes, int n_in,
                              void* d_out, int out_size, void* d_ws, size_t ws_size,
                              hipStream_t stream) {
  const float* X      = (const float*)d_in[0];
  const float* adj    = (const float*)d_in[1];
  const float* Wrel0  = (const float*)d_in[2];
  const float* brel0  = (const float*)d_in[3];
  const float* Wroot0 = (const float*)d_in[4];
  const float* Wrel1  = (const float*)d_in[5];
  const float* brel1  = (const float*)d_in[6];
  const float* Wroot1 = (const float*)d_in[7];
  const float* Wrel2  = (const float*)d_in[8];
  const float* brel2  = (const float*)d_in[9];
  const float* Wroot2 = (const float*)d_in[10];
  float* out = (float*)d_out;

  char* ws = (char*)d_ws;
  __bf16* aggbf = (__bf16*)ws;  ws += (size_t)NNODES * 256 * sizeof(__bf16);
  __bf16* xbf   = (__bf16*)ws;  ws += (size_t)NNODES * 256 * sizeof(__bf16);
  __bf16* h1bf  = (__bf16*)ws;  ws += (size_t)NNODES * 64 * sizeof(__bf16);
  __bf16* h2bf  = (__bf16*)ws;  ws += (size_t)NNODES * 64 * sizeof(__bf16);
  __bf16* wr0 = (__bf16*)ws;    ws += 64 * 256 * sizeof(__bf16);
  __bf16* wo0 = (__bf16*)ws;    ws += 64 * 256 * sizeof(__bf16);
  __bf16* wr1 = (__bf16*)ws;    ws += 64 * 64 * sizeof(__bf16);
  __bf16* wo1 = (__bf16*)ws;    ws += 64 * 64 * sizeof(__bf16);
  __bf16* wr2 = (__bf16*)ws;    ws += 64 * 64 * sizeof(__bf16);
  __bf16* wo2 = (__bf16*)ws;    ws += 64 * 64 * sizeof(__bf16);

  auto cvt = [&](const float* s, __bf16* d, int n) {
    int n4 = n / 4;
    cvt_f32_to_bf16<<<(n4 + 255) / 256, 256, 0, stream>>>(s, d, n4);
  };

  // One-time conversions.
  cvt(Wrel0, wr0, 64 * 256);
  cvt(Wroot0, wo0, 64 * 256);
  cvt(Wrel1, wr1, 64 * 64);
  cvt(Wroot1, wo1, 64 * 64);
  cvt(Wrel2, wr2, 64 * 64);
  cvt(Wroot2, wo2, 64 * 64);
  cvt(X, xbf, NNODES * 256);

  // Layer 0 (Fin = 256)
  gemm_aggT<<<dim3(NNODES / BM, 256 / BN), 256, 0, stream>>>(adj, xbf, aggbf, 256);
  combine_relu<false><<<NNODES / 16 / 8, 256, 0, stream>>>(aggbf, xbf, wr0, wo0, brel0,
                                                           h1bf, 256);
  // Layer 1 (Fin = 64)
  gemm_aggT<<<dim3(NNODES / BM, 1), 256, 0, stream>>>(adj, h1bf, aggbf, 64);
  combine_relu<false><<<NNODES / 16 / 8, 256, 0, stream>>>(aggbf, h1bf, wr1, wo1, brel1,
                                                           h2bf, 64);
  // Layer 2 (Fin = 64) -> fp32 d_out
  gemm_aggT<<<dim3(NNODES / BM, 1), 256, 0, stream>>>(adj, h2bf, aggbf, 64);
  combine_relu<true><<<NNODES / 16 / 8, 256, 0, stream>>>(aggbf, h2bf, wr2, wo2, brel2,
                                                          out, 64);
}
